// EncoderStack_81449759801620
// MI455X (gfx1250) — compile-verified, matrix operand order checked by reading, not run on hardware
//
#include <hip/hip_runtime.h>
#include <hip/hip_bf16.h>

typedef _Float16 half_t;
typedef __attribute__((ext_vector_type(16))) _Float16 v16h;
typedef __attribute__((ext_vector_type(8)))  float    v8f;

// Transformer dims (fixed by reference)
#define BB 4
#define TT 1024
#define DD 1024
#define HH 16
#define DKK 64
#define DFF 4096
#define LDSS 40   // padded LDS row stride in halves (80B, dword-aligned)

union Frag { v16h v; unsigned u[8]; half_t h[16]; };

// K index for element-pair v (VGPR v) of a 16-bit A/B fragment, per CDNA5 ISA
// 16-bit A-matrix 16x32 layout: lanes 0-15 -> +0, lanes 16-31 -> +8.
__device__ __forceinline__ int kmap(int v, int hi) {
    int kb = (v < 4) ? (2 * v) : (16 + 2 * (v - 4));
    return kb + (hi ? 8 : 0);
}

__device__ __forceinline__ v8f wmma16(v16h a, v16h b, v8f c) {
    return __builtin_amdgcn_wmma_f32_16x16x32_f16(
        /*neg_a=*/false, a, /*neg_b=*/false, b,
        /*c_mod=*/(short)0, c, /*reuse_a=*/false, /*reuse_b=*/false);
}

// ---------------------------------------------------------------------------
// fp32 -> fp16 elementwise conversion
// ---------------------------------------------------------------------------
__global__ __launch_bounds__(256)
void f32_to_f16_kernel(const float* __restrict__ in, half_t* __restrict__ out, long n) {
    long i = (long)blockIdx.x * 256 + threadIdx.x;
    if (i < n) out[i] = (half_t)in[i];
}

// ---------------------------------------------------------------------------
// WMMA GEMM: C[M,N] = A[M,K] @ B[K,N] (+bias) (relu?) ; A,B f16, f32 accum.
// Block: 256 threads = 8 waves; block tile 128x128; wave tile 32x64
// (2 A-frags x 4 B-frags -> 8 WMMAs per 32-wide K step).
// M,N multiples of 128; K multiple of 32 (true for all uses here).
// ---------------------------------------------------------------------------
template<bool BIAS, bool RELU, bool OUT16>
__global__ __launch_bounds__(256)
void gemm_wmma_kernel(const half_t* __restrict__ A, const half_t* __restrict__ B,
                      const float* __restrict__ bias,
                      half_t* __restrict__ C16, float* __restrict__ C32,
                      int M, int N, int K) {
    __shared__ half_t As[128 * LDSS];   // [row m][k]
    __shared__ half_t Bs[128 * LDSS];   // stored transposed: [col n][k]

    const int tid  = threadIdx.x;
    const int lane = tid & 31;
    const int wave = tid >> 5;
    const int hi   = (lane >= 16) ? 1 : 0;
    const int l16  = lane & 15;
    const int m0   = blockIdx.y * 128;
    const int n0   = blockIdx.x * 128;
    const int wm   = (wave >> 1) * 32;  // 0,32,64,96
    const int wn   = (wave & 1) * 64;   // 0,64

    v8f acc[2][4] = {};

    for (int kt = 0; kt < K; kt += 32) {
        // --- load A tile (128x32 halves) as dwords, coalesced ---
#pragma unroll
        for (int i = 0; i < 8; ++i) {
            int idx = i * 256 + tid;          // 0..2047 dwords
            int row = idx >> 4;
            int cp  = idx & 15;               // dword (pair) index within row
            const unsigned* src =
                (const unsigned*)(A + (long)(m0 + row) * K + kt + cp * 2);
            *(unsigned*)(&As[row * LDSS + cp * 2]) = *src;
        }
        // --- load B tile (32x128) transposed into Bs[n][k] ---
#pragma unroll
        for (int i = 0; i < 16; ++i) {
            int idx = i * 256 + tid;          // 0..4095 halves
            int k = idx >> 7;
            int n = idx & 127;
            Bs[n * LDSS + k] = B[(long)(kt + k) * N + n0 + n];
        }
        if (kt + 32 < K) {
            // hint the next A tile toward the caches
            __builtin_prefetch(A + (long)(m0 + (tid >> 1)) * K + kt + 32, 0, 1);
        }
        __syncthreads();

        Frag af[2], bf[4];
#pragma unroll
        for (int f = 0; f < 2; ++f) {
            const half_t* ar = &As[(wm + f * 16 + l16) * LDSS];
#pragma unroll
            for (int v = 0; v < 8; ++v)
                af[f].u[v] = *(const unsigned*)(ar + kmap(v, hi));
        }
#pragma unroll
        for (int j = 0; j < 4; ++j) {
            const half_t* br = &Bs[(wn + j * 16 + l16) * LDSS];
#pragma unroll
            for (int v = 0; v < 8; ++v)
                bf[j].u[v] = *(const unsigned*)(br + kmap(v, hi));
        }
#pragma unroll
        for (int i = 0; i < 2; ++i)
#pragma unroll
            for (int j = 0; j < 4; ++j)
                acc[i][j] = wmma16(af[i].v, bf[j].v, acc[i][j]);
        __syncthreads();
    }

    // --- epilogue: C layout -> lane l holds row (r + 8*hi), col l16 ---
#pragma unroll
    for (int i = 0; i < 2; ++i) {
#pragma unroll
        for (int j = 0; j < 4; ++j) {
            int n = n0 + wn + j * 16 + l16;
            float bv = BIAS ? bias[n] : 0.0f;
#pragma unroll
            for (int r = 0; r < 8; ++r) {
                int m = m0 + wm + i * 16 + r + hi * 8;
                float val = acc[i][j][r] + bv;
                if (RELU) val = fmaxf(val, 0.0f);
                if (OUT16) C16[(long)m * N + n] = (half_t)val;
                else       C32[(long)m * N + n] = val;
            }
        }
    }
}

// ---------------------------------------------------------------------------
// Flash attention: one wave handles one 16-row Q block of one (b,h).
// Q,K,V are f16 in (B,T,D) layout with head h at cols [h*64, h*64+64).
// Streaming softmax over key blocks of 32; output f32 to att (B,T,D).
// ---------------------------------------------------------------------------
__global__ __launch_bounds__(32)
void attn_kernel(const half_t* __restrict__ Q, const half_t* __restrict__ Km,
                 const half_t* __restrict__ V, float* __restrict__ att) {
    __shared__ half_t pl[16 * LDSS];   // P tile, [row m][k]

    const int lane = threadIdx.x & 31;
    const int hi   = (lane >= 16) ? 1 : 0;
    const int l16  = lane & 15;
    const int b  = blockIdx.z;
    const int h  = blockIdx.y;
    const int qb = blockIdx.x;          // q block index (16 rows)

    // Q fragments (A-layout): q[0] covers contraction 0..31, q[1] 32..63
    Frag q[2];
    {
        const long baseQ = (long)(b * TT + qb * 16 + l16) * DD + h * DKK;
#pragma unroll
        for (int f = 0; f < 2; ++f)
#pragma unroll
            for (int v = 0; v < 8; ++v)
                q[f].u[v] = *(const unsigned*)(Q + baseQ + f * 32 + kmap(v, hi));
    }

    float mi[8], li[8];
    v8f o[4] = {};
#pragma unroll
    for (int r = 0; r < 8; ++r) { mi[r] = -1e30f; li[r] = 0.0f; }

    for (int j = 0; j < TT; j += 32) {
        // K fragments (B-layout == same gather as A from row-major K matrix)
        Frag kf[2][2];
#pragma unroll
        for (int s = 0; s < 2; ++s) {
            const long baseK = (long)(b * TT + j + s * 16 + l16) * DD + h * DKK;
#pragma unroll
            for (int f = 0; f < 2; ++f)
#pragma unroll
                for (int v = 0; v < 8; ++v)
                    kf[s][f].u[v] = *(const unsigned*)(Km + baseK + f * 32 + kmap(v, hi));
        }
        v8f c0 = {}, c1 = {};
        c0 = wmma16(q[0].v, kf[0][0].v, c0);
        c0 = wmma16(q[1].v, kf[0][1].v, c0);
        c1 = wmma16(q[0].v, kf[1][0].v, c1);
        c1 = wmma16(q[1].v, kf[1][1].v, c1);

        // streaming softmax update (row = r + 8*hi, spread across 16 lanes)
        float p0[8], p1[8];
#pragma unroll
        for (int r = 0; r < 8; ++r) {
            float s0 = c0[r] * 0.125f;     // 1/sqrt(64)
            float s1 = c1[r] * 0.125f;
            float mx = fmaxf(s0, s1);
#pragma unroll
            for (int d = 1; d < 16; d <<= 1) mx = fmaxf(mx, __shfl_xor(mx, d, 32));
            float mnew  = fmaxf(mi[r], mx);
            float alpha = __expf(mi[r] - mnew);
            float e0 = __expf(s0 - mnew);
            float e1 = __expf(s1 - mnew);
            float rs = e0 + e1;
#pragma unroll
            for (int d = 1; d < 16; d <<= 1) rs += __shfl_xor(rs, d, 32);
            li[r] = li[r] * alpha + rs;
            mi[r] = mnew;
            p0[r] = e0; p1[r] = e1;
#pragma unroll
            for (int c = 0; c < 4; ++c) o[c][r] *= alpha;
        }

        // P (f16) to LDS in row-major [m][k], then reload as A fragment
#pragma unroll
        for (int r = 0; r < 8; ++r) {
            int m = r + hi * 8;
            pl[m * LDSS + l16]      = (half_t)p0[r];
            pl[m * LDSS + 16 + l16] = (half_t)p1[r];
        }
        __syncthreads();
        Frag pa;
        {
            const half_t* pr = &pl[l16 * LDSS];
#pragma unroll
            for (int v = 0; v < 8; ++v)
                pa.u[v] = *(const unsigned*)(pr + kmap(v, hi));
        }
        // V B-fragments gathered from global (strided), 4 output col chunks
#pragma unroll
        for (int c = 0; c < 4; ++c) {
            Frag vb;
            const long baseV = (long)(b * TT + j) * DD + h * DKK + c * 16 + l16;
#pragma unroll
            for (int v = 0; v < 8; ++v) {
                int kk = kmap(v, hi);
                vb.h[2 * v]     = V[baseV + (long)kk * DD];
                vb.h[2 * v + 1] = V[baseV + (long)(kk + 1) * DD];
            }
            o[c] = wmma16(pa.v, vb.v, o[c]);
        }
        __syncthreads();
    }

    // normalize and store (f32)
#pragma unroll
    for (int r = 0; r < 8; ++r) {
        float inv = 1.0f / li[r];
#pragma unroll
        for (int c = 0; c < 4; ++c) {
            long idx = (long)(b * TT + qb * 16 + r + hi * 8) * DD + h * DKK + c * 16 + l16;
            att[idx] = o[c][r] * inv;
        }
    }
}

// ---------------------------------------------------------------------------
// Fused residual + LayerNorm over D=1024; block = 256 threads, one row each.
// Optionally also emits the f16 copy for the following GEMM.
// ---------------------------------------------------------------------------
template<bool W16>
__global__ __launch_bounds__(256)
void ln_residual_kernel(const float* __restrict__ x, const float* __restrict__ y,
                        const float* __restrict__ g, const float* __restrict__ be,
                        float* __restrict__ out32, half_t* __restrict__ out16) {
    __shared__ float sred[256];
    const int tid = threadIdx.x;
    const long row = blockIdx.x;
    const float* xr = x + row * DD;
    const float* yr = y + row * DD;

    float v[4];
    float s = 0.0f;
#pragma unroll
    for (int i = 0; i < 4; ++i) {
        int idx = tid + i * 256;
        v[i] = xr[idx] + yr[idx];
        s += v[i];
    }
    sred[tid] = s; __syncthreads();
    for (int off = 128; off > 0; off >>= 1) {
        if (tid < off) sred[tid] += sred[tid + off];
        __syncthreads();
    }
    float mu = sred[0] * (1.0f / DD);
    __syncthreads();

    float s2 = 0.0f;
#pragma unroll
    for (int i = 0; i < 4; ++i) { float d = v[i] - mu; s2 += d * d; }
    sred[tid] = s2; __syncthreads();
    for (int off = 128; off > 0; off >>= 1) {
        if (tid < off) sred[tid] += sred[tid + off];
        __syncthreads();
    }
    float rstd = rsqrtf(sred[0] * (1.0f / DD) + 1e-5f);

#pragma unroll
    for (int i = 0; i < 4; ++i) {
        int idx = tid + i * 256;
        float val = (v[i] - mu) * rstd * g[idx] + be[idx];
        out32[row * DD + idx] = val;
        if (W16) out16[row * DD + idx] = (half_t)val;
    }
}

// ---------------------------------------------------------------------------
extern "C" void kernel_launch(void* const* d_in, const int* in_sizes, int n_in,
                              void* d_out, int out_size, void* d_ws, size_t ws_size,
                              hipStream_t stream) {
    (void)in_sizes; (void)n_in; (void)out_size; (void)ws_size;

    const float* x     = (const float*)d_in[0];
    const float* Wq    = (const float*)d_in[1];
    const float* Wk    = (const float*)d_in[2];
    const float* Wv    = (const float*)d_in[3];
    const float* W1    = (const float*)d_in[4];
    const float* b1    = (const float*)d_in[5];
    const float* W2    = (const float*)d_in[6];
    const float* b2    = (const float*)d_in[7];
    const float* ln1g  = (const float*)d_in[8];
    const float* ln1b  = (const float*)d_in[9];
    const float* ln2g  = (const float*)d_in[10];
    const float* ln2b  = (const float*)d_in[11];

    const long M = (long)BB * TT;          // 4096 rows

    // scratch carve-out
    size_t off = 0;
    auto alloc = [&](size_t bytes) {
        void* p = (char*)d_ws + off;
        off += (bytes + 255) & ~(size_t)255;
        return p;
    };
    half_t* xh   = (half_t*)alloc(M * DD * 2);
    half_t* wqh  = (half_t*)alloc((size_t)DD * DD * 2);
    half_t* wkh  = (half_t*)alloc((size_t)DD * DD * 2);
    half_t* wvh  = (half_t*)alloc((size_t)DD * DD * 2);
    half_t* w1h  = (half_t*)alloc((size_t)DD * DFF * 2);
    half_t* w2h  = (half_t*)alloc((size_t)DFF * DD * 2);
    half_t* Qh   = (half_t*)alloc(M * DD * 2);
    half_t* Kh   = (half_t*)alloc(M * DD * 2);
    half_t* Vh   = (half_t*)alloc(M * DD * 2);
    float*  att  = (float*) alloc(M * DD * 4);
    float*  x1   = (float*) alloc(M * DD * 4);
    half_t* x1h  = (half_t*)alloc(M * DD * 2);
    half_t* hh   = (half_t*)alloc(M * DFF * 2);
    float*  ff   = (float*) alloc(M * DD * 4);

    // 1) precision conversions
    f32_to_f16_kernel<<<(M * DD) / 256, 256, 0, stream>>>(x,  xh,  M * DD);
    f32_to_f16_kernel<<<(DD * DD) / 256, 256, 0, stream>>>(Wq, wqh, (long)DD * DD);
    f32_to_f16_kernel<<<(DD * DD) / 256, 256, 0, stream>>>(Wk, wkh, (long)DD * DD);
    f32_to_f16_kernel<<<(DD * DD) / 256, 256, 0, stream>>>(Wv, wvh, (long)DD * DD);
    f32_to_f16_kernel<<<(DD * DFF) / 256, 256, 0, stream>>>(W1, w1h, (long)DD * DFF);
    f32_to_f16_kernel<<<(DFF * DD) / 256, 256, 0, stream>>>(W2, w2h, (long)DFF * DD);

    // 2) Q/K/V projections (WMMA GEMMs, f16 out)
    dim3 gQKV(DD / 128, M / 128);
    gemm_wmma_kernel<false, false, true><<<gQKV, 256, 0, stream>>>(
        xh, wqh, nullptr, Qh, nullptr, (int)M, DD, DD);
    gemm_wmma_kernel<false, false, true><<<gQKV, 256, 0, stream>>>(
        xh, wkh, nullptr, Kh, nullptr, (int)M, DD, DD);
    gemm_wmma_kernel<false, false, true><<<gQKV, 256, 0, stream>>>(
        xh, wvh, nullptr, Vh, nullptr, (int)M, DD, DD);

    // 3) flash attention
    attn_kernel<<<dim3(TT / 16, HH, BB), 32, 0, stream>>>(Qh, Kh, Vh, att);

    // 4) x1 = LN(x + att), also f16 copy
    ln_residual_kernel<true><<<M, 256, 0, stream>>>(x, att, ln1g, ln1b, x1, x1h);

    // 5) FFN: hh = relu(x1 @ W1 + b1)   (f16 out)
    gemm_wmma_kernel<true, true, true><<<dim3(DFF / 128, M / 128), 256, 0, stream>>>(
        x1h, w1h, b1, hh, nullptr, (int)M, DFF, DD);

    // 6) ff = hh @ W2 + b2              (f32 out)
    gemm_wmma_kernel<true, false, false><<<dim3(DD / 128, M / 128), 256, 0, stream>>>(
        hh, w2h, b2, nullptr, ff, (int)M, DD, DFF);

    // 7) out = LN(x1 + ff)
    ln_residual_kernel<false><<<M, 256, 0, stream>>>(x1, ff, ln2g, ln2b,
                                                     (float*)d_out, nullptr);
}
